// MoEBlock_33389075759482
// MI455X (gfx1250) — matrix operand order repaired and checked by reading, not compiled, not run on hardware
//
#include <hip/hip_runtime.h>

// ---------------- CDNA5 WMMA types ----------------
typedef __attribute__((ext_vector_type(16))) _Float16 v16h;
typedef __attribute__((ext_vector_type(8)))  float    v8f;

constexpr int T = 4096;   // tokens
constexpr int D = 1024;   // model dim
constexpr int H = 4096;   // hidden dim
constexpr int E = 8;      // experts
// Max padded rows: sum expert counts = 2T = 8192 -> padded <= 9088, + shared T=4096
constexpr int RCAP   = 13312;
constexpr int MTILES = RCAP / 128;   // 104

// Async global->LDS copy via inline asm (bypasses builtin signature issues).
#define USE_ASYNC_ASM 1

static __device__ __forceinline__ unsigned short f2h(float f) {
  union { _Float16 h; unsigned short u; } c; c.h = (_Float16)f; return c.u;
}
static __device__ __forceinline__ unsigned pack2(float a, float b) {
  return (unsigned)f2h(a) | ((unsigned)f2h(b) << 16);
}
static __device__ __forceinline__ float gelu_erf(float v) {
  return 0.5f * v * (1.0f + erff(v * 0.70710678118654752f));
}

// Convert 16 f32 -> f16 and store one A-stage slice (two 16B LDS stores)
static __device__ __forceinline__ void storeA(unsigned short (*As)[40], int ar, int ac,
                                              float4 f0, float4 f1, float4 f2, float4 f3) {
  uint4 q0 = make_uint4(pack2(f0.x,f0.y), pack2(f0.z,f0.w), pack2(f1.x,f1.y), pack2(f1.z,f1.w));
  uint4 q1 = make_uint4(pack2(f2.x,f2.y), pack2(f2.z,f2.w), pack2(f3.x,f3.y), pack2(f3.z,f3.w));
  *(uint4*)&As[ar][ac]     = q0;
  *(uint4*)&As[ar][ac + 8] = q1;
}
// Convert 16 f32 -> f16 and store transposed B-stage slice
static __device__ __forceinline__ void storeB(unsigned short (*Bs)[40], int bn, int bk,
                                              float4 g0, float4 g1, float4 g2, float4 g3) {
  float v[16] = {g0.x,g0.y,g0.z,g0.w, g1.x,g1.y,g1.z,g1.w,
                 g2.x,g2.y,g2.z,g2.w, g3.x,g3.y,g3.z,g3.w};
#pragma unroll
  for (int i = 0; i < 16; ++i) Bs[bn + i][bk] = f2h(v[i]);
}

// ---------------- Gate: softmax + top-2 -> dense combine weights ----------------
// One wave per token: x row read once, 8 gate dots in parallel per lane.
__global__ void gate_kernel(const float* __restrict__ x, const float* __restrict__ gw,
                            float* __restrict__ w_tok) {
  int wid = threadIdx.x >> 5, lane = threadIdx.x & 31;
  int t = blockIdx.x * 8 + wid;
  const float* xr = x + (size_t)t * D;
  float acc[E] = {0.f, 0.f, 0.f, 0.f, 0.f, 0.f, 0.f, 0.f};
  for (int d = lane; d < D; d += 32) {
    float xv = xr[d];
#pragma unroll
    for (int e = 0; e < E; ++e) acc[e] += xv * gw[(size_t)e * D + d];
  }
#pragma unroll
  for (int e = 0; e < E; ++e)
#pragma unroll
    for (int off = 16; off > 0; off >>= 1) acc[e] += __shfl_xor(acc[e], off, 32);
  if (lane == 0) {
    float m = acc[0];
    for (int e = 1; e < E; ++e) m = fmaxf(m, acc[e]);
    float p[E], s = 0.f;
    for (int e = 0; e < E; ++e) { p[e] = expf(acc[e] - m); s += p[e]; }
    float inv = 1.f / s;
    for (int e = 0; e < E; ++e) p[e] *= inv;
    int i1 = 0;
    for (int e = 1; e < E; ++e) if (p[e] > p[i1]) i1 = e;
    int i2 = (i1 == 0) ? 1 : 0;
    for (int e = 0; e < E; ++e) if (e != i1 && p[e] > p[i2]) i2 = e;
    float* wr = w_tok + (size_t)t * E;
    for (int e = 0; e < E; ++e) wr[e] = (e == i1 || e == i2) ? p[e] : 0.f;
  }
}

// ---------------- Deterministic row-list build (ballot prefix scan) ----------------
__global__ void build_kernel(const float* __restrict__ w_tok, int* __restrict__ segoff,
                             int* __restrict__ tileseg, int* __restrict__ rowtok,
                             float* __restrict__ roww) {
  __shared__ int cnt[E];
  __shared__ int off[E + 2];
  int wid = threadIdx.x >> 5, lane = threadIdx.x & 31;
  if (wid < E) {
    int c = 0;
    for (int t0 = 0; t0 < T; t0 += 32) {
      float w = w_tok[(size_t)(t0 + lane) * E + wid];
      unsigned long long m = __ballot(w > 0.f);
      c += __popc((unsigned)m);
    }
    if (lane == 0) cnt[wid] = c;
  }
  __syncthreads();
  if (threadIdx.x == 0) {
    off[0] = 0;
    for (int e = 0; e < E; ++e) off[e + 1] = off[e] + ((cnt[e] + 127) & ~127);
    off[E + 1] = off[E] + T;                    // shared segment (all tokens)
    for (int i = 0; i < E + 2; ++i) segoff[i] = off[i];
    for (int tile = 0; tile < MTILES; ++tile) {
      int r = tile * 128, s = -1;
      for (int q = 0; q <= E; ++q) if (r >= off[q] && r < off[q + 1]) s = q;
      tileseg[tile] = s;
    }
  }
  __syncthreads();
  if (wid < E) {
    int base = off[wid], pos = 0;
    for (int t0 = 0; t0 < T; t0 += 32) {
      int t = t0 + lane;
      float w = w_tok[(size_t)t * E + wid];
      unsigned long long m64 = __ballot(w > 0.f);
      unsigned m = (unsigned)m64;
      if (w > 0.f) {
        int r = base + pos + __popc(m & ((1u << lane) - 1u));
        rowtok[r] = t; roww[r] = w;
      }
      pos += __popc(m);
    }
    for (int r = base + cnt[wid] + lane; r < off[wid + 1]; r += 32) {
      rowtok[r] = 0; roww[r] = 0.f;             // padded rows: compute, never stored
    }
  }
  __syncthreads();
  int sbase = off[E];
  for (int r = (int)threadIdx.x; r < T; r += (int)blockDim.x) {
    rowtok[sbase + r] = r; roww[sbase + r] = 1.0f;   // shared expert, weight 1
  }
}

// ---------------- GEMM1: h = gelu(gather(x) @ w_in + b_in), double-buffered ----------------
__global__ __launch_bounds__(256) void gemm1_kernel(
    const float* __restrict__ x, const float* __restrict__ w_in, const float* __restrict__ sw_in,
    const float* __restrict__ b_in, const float* __restrict__ sb_in,
    const int* __restrict__ rowtok, const int* __restrict__ tileseg,
    unsigned short* __restrict__ h) {
  int bx = blockIdx.x, by = blockIdx.y;
  int seg = tileseg[bx];
  if (seg < 0) return;
  const float* W    = (seg < E) ? (w_in + (size_t)seg * D * H) : sw_in;   // [D,H] row-major
  const float* Bias = (seg < E) ? (b_in + (size_t)seg * H) : sb_in;
  const int n0 = by * 128;

  __shared__ __align__(16) unsigned short As[2][128][40];
  __shared__ __align__(16) unsigned short Bs[2][128][40];
  __shared__ int toks[128];

  int tid = threadIdx.x;
  if (tid < 128) toks[tid] = rowtok[bx * 128 + tid];
  __syncthreads();

  int wid = tid >> 5, lane = tid & 31;
  int wm = wid >> 2, wn = wid & 3;          // 2x4 wave grid: 64x32 per wave
  int lrow = lane & 15, lhi = lane >> 4;

  v8f acc[4][2];
  v8f zero = {0.f, 0.f, 0.f, 0.f, 0.f, 0.f, 0.f, 0.f};
#pragma unroll
  for (int f = 0; f < 4; ++f)
#pragma unroll
    for (int g = 0; g < 2; ++g) acc[f][g] = zero;

  int ar = tid >> 1, ac = (tid & 1) * 16;   // A stage: 2 threads/row, 16 cols each
  int bk = tid >> 3, bn = (tid & 7) * 16;   // B stage: 8 threads/K-row, 16 cols each
  const float* asrc = x + (size_t)toks[ar] * D + ac;     // +k0
  const float* bsrc = W + (size_t)bk * H + (n0 + bn);    // +k0*H

  // Prologue: stage tile 0 into buffer 0
  {
    float4 a0 = *(const float4*)(asrc + 0), a1 = *(const float4*)(asrc + 4);
    float4 a2 = *(const float4*)(asrc + 8), a3 = *(const float4*)(asrc + 12);
    float4 g0 = *(const float4*)(bsrc + 0), g1 = *(const float4*)(bsrc + 4);
    float4 g2 = *(const float4*)(bsrc + 8), g3 = *(const float4*)(bsrc + 12);
    storeA(As[0], ar, ac, a0, a1, a2, a3);
    storeB(Bs[0], bn, bk, g0, g1, g2, g3);
  }
  __syncthreads();

  const int NK = D / 32;
  for (int it = 0; it < NK; ++it) {
    int cur = it & 1, nxt = cur ^ 1;
    bool more = (it + 1) < NK;
    float4 a0, a1, a2, a3, g0, g1, g2, g3;
    if (more) {   // issue next tile's global loads before compute (latency overlap)
      const float* ap = asrc + (it + 1) * 32;
      a0 = *(const float4*)(ap + 0);  a1 = *(const float4*)(ap + 4);
      a2 = *(const float4*)(ap + 8);  a3 = *(const float4*)(ap + 12);
      const float* bp = bsrc + (size_t)(it + 1) * 32 * H;
      g0 = *(const float4*)(bp + 0);  g1 = *(const float4*)(bp + 4);
      g2 = *(const float4*)(bp + 8);  g3 = *(const float4*)(bp + 12);
    }

    // Compute on current buffer
    v16h a[4], b[2];
#pragma unroll
    for (int f = 0; f < 4; ++f) {      // A frag: K = kb..kb+15 contiguous, kb = lhi*16
      int row = wm * 64 + f * 16 + lrow;
      int kb = lhi * 16;
      union { uint4 q[2]; v16h v; } u;
      u.q[0] = *(const uint4*)&As[cur][row][kb];
      u.q[1] = *(const uint4*)&As[cur][row][kb + 8];
      a[f] = u.v;
    }
#pragma unroll
    for (int g = 0; g < 2; ++g) {      // B frag: K = kb..kb+15 contiguous, kb = lhi*16
      int n = wn * 32 + g * 16 + lrow;
      int kb = lhi * 16;
      union { uint4 q[2]; v16h v; } u;
      u.q[0] = *(const uint4*)&Bs[cur][n][kb];
      u.q[1] = *(const uint4*)&Bs[cur][n][kb + 8];
      b[g] = u.v;
    }
#pragma unroll
    for (int f = 0; f < 4; ++f)
#pragma unroll
      for (int g = 0; g < 2; ++g)
        acc[f][g] = __builtin_amdgcn_wmma_f32_16x16x32_f16(
            false, a[f], false, b[g], (short)0, acc[f][g], false, false);

    if (more) {   // convert + publish next buffer
      storeA(As[nxt], ar, ac, a0, a1, a2, a3);
      storeB(Bs[nxt], bn, bk, g0, g1, g2, g3);
    }
    __syncthreads();
  }

  // Epilogue: +bias, exact-erf GELU, store f16 h
#pragma unroll
  for (int g = 0; g < 2; ++g) {
    int n = n0 + wn * 32 + g * 16 + lrow;
    float bias = Bias[n];
#pragma unroll
    for (int f = 0; f < 4; ++f) {
#pragma unroll
      for (int i = 0; i < 8; ++i) {
        int ml = wm * 64 + f * 16 + lhi * 8 + i;   // C layout: VGPR i -> M = i + 8*lhi
        float v = acc[f][g][i] + bias;
        h[(size_t)(bx * 128 + ml) * H + n] = f2h(gelu_erf(v));
      }
    }
  }
}

// ---------------- GEMM2: y[token] += w * (h @ w_out + b_out), per-segment ----------------
__global__ __launch_bounds__(256) void gemm2_kernel(
    const unsigned short* __restrict__ h, const float* __restrict__ w_out,
    const float* __restrict__ sw_out, const float* __restrict__ b_out,
    const float* __restrict__ sb_out, const int* __restrict__ segoff,
    const int* __restrict__ rowtok, const float* __restrict__ roww,
    float* __restrict__ y, int s) {
  int off0 = segoff[s], off1 = segoff[s + 1];
  int m0 = off0 + blockIdx.x * 128;
  if (m0 >= off1) return;
  const float* W    = (s < E) ? (w_out + (size_t)s * H * D) : sw_out;   // [H,D] row-major
  const float* Bias = (s < E) ? (b_out + (size_t)s * D) : sb_out;
  int n0 = blockIdx.y * 128;

  __shared__ __align__(16) unsigned short As[2][128][40];
  __shared__ __align__(16) unsigned short Bs[2][128][40];
  __shared__ int   rt[128];
  __shared__ float rw[128];

  int tid = threadIdx.x;
  if (tid < 128) { rt[tid] = rowtok[m0 + tid]; rw[tid] = roww[m0 + tid]; }
  __syncthreads();

  int wid = tid >> 5, lane = tid & 31;
  int wm = wid >> 2, wn = wid & 3;
  int lrow = lane & 15, lhi = lane >> 4;

  v8f acc[4][2];
  v8f zero = {0.f, 0.f, 0.f, 0.f, 0.f, 0.f, 0.f, 0.f};
#pragma unroll
  for (int f = 0; f < 4; ++f)
#pragma unroll
    for (int g = 0; g < 2; ++g) acc[f][g] = zero;

  int ar = tid >> 1, ac = (tid & 1) * 16;
  int bk = tid >> 3, bn = (tid & 7) * 16;
  const unsigned short* asrc = h + (size_t)(m0 + ar) * H + ac;   // +k0 (f16 elements)
  const float* bsrc = W + (size_t)bk * D + (n0 + bn);            // +k0*D

  // A tile copy: f16 bytes straight to LDS.
  // ASYNC path: GLOBAL_LOAD_ASYNC_TO_LDS_B128 via inline asm (VDST = LDS address VGPR,
  // VADDR = 64-bit global address; per ISA 10.2 the low 32 bits of a flat LDS pointer
  // are the LDS address). Tracked by ASYNCcnt; drained with s_wait_asynccnt before the
  // publishing barrier.
#if USE_ASYNC_ASM
  #define STAGE_A(buf, it)                                                              \
    do {                                                                                \
      unsigned loff = (unsigned)(size_t)&As[(buf)][ar][ac];                             \
      const unsigned short* gp = asrc + (size_t)(it) * 32;                              \
      asm volatile("global_load_async_to_lds_b128 %0, %1, off\n\t"                      \
                   "global_load_async_to_lds_b128 %0, %1, off offset:16"                \
                   :: "v"(loff), "v"(gp) : "memory");                                   \
    } while (0)
  #define WAIT_ASYNC() asm volatile("s_wait_asynccnt 0x0" ::: "memory")
#else
  #define STAGE_A(buf, it)                                                              \
    do {                                                                                \
      const unsigned short* sp = asrc + (size_t)(it) * 32;                              \
      uint4 q0 = *(const uint4*)(sp);                                                   \
      uint4 q1 = *(const uint4*)(sp + 8);                                               \
      *(uint4*)&As[(buf)][ar][ac]     = q0;                                             \
      *(uint4*)&As[(buf)][ar][ac + 8] = q1;                                             \
    } while (0)
  #define WAIT_ASYNC() do {} while (0)
#endif

  // Prologue: stage tile 0 into buffer 0
  {
    STAGE_A(0, 0);
    float4 g0 = *(const float4*)(bsrc + 0), g1 = *(const float4*)(bsrc + 4);
    float4 g2 = *(const float4*)(bsrc + 8), g3 = *(const float4*)(bsrc + 12);
    storeB(Bs[0], bn, bk, g0, g1, g2, g3);
    WAIT_ASYNC();
  }
  __syncthreads();

  const int NK = H / 32;
  for (int it = 0; it < NK; ++it) {
    int cur = it & 1, nxt = cur ^ 1;
    bool more = (it + 1) < NK;
    float4 g0, g1, g2, g3;
    if (more) {
      STAGE_A(nxt, it + 1);   // async: fire-and-forget into next buffer
      const float* bp = bsrc + (size_t)(it + 1) * 32 * D;
      g0 = *(const float4*)(bp + 0);  g1 = *(const float4*)(bp + 4);
      g2 = *(const float4*)(bp + 8);  g3 = *(const float4*)(bp + 12);
    }

    v16h a[4], b[2];
#pragma unroll
    for (int f = 0; f < 4; ++f) {
      int row = wm * 64 + f * 16 + lrow;
      int kb = lhi * 16;
      union { uint4 q[2]; v16h v; } u;
      u.q[0] = *(const uint4*)&As[cur][row][kb];
      u.q[1] = *(const uint4*)&As[cur][row][kb + 8];
      a[f] = u.v;
    }
#pragma unroll
    for (int g = 0; g < 2; ++g) {
      int n = wn * 32 + g * 16 + lrow;
      int kb = lhi * 16;
      union { uint4 q[2]; v16h v; } u;
      u.q[0] = *(const uint4*)&Bs[cur][n][kb];
      u.q[1] = *(const uint4*)&Bs[cur][n][kb + 8];
      b[g] = u.v;
    }
#pragma unroll
    for (int f = 0; f < 4; ++f)
#pragma unroll
      for (int g = 0; g < 2; ++g)
        acc[f][g] = __builtin_amdgcn_wmma_f32_16x16x32_f16(
            false, a[f], false, b[g], (short)0, acc[f][g], false, false);

    if (more) storeB(Bs[nxt], bn, bk, g0, g1, g2, g3);
    WAIT_ASYNC();
    __syncthreads();
  }

  // Epilogue: weighted accumulate into y[token]; padded rows (w==0) skipped.
  // No races: within one segment each token appears at most once; segments serialized.
#pragma unroll
  for (int g = 0; g < 2; ++g) {
    int n = n0 + wn * 32 + g * 16 + lrow;
    float bias = Bias[n];
#pragma unroll
    for (int f = 0; f < 4; ++f) {
#pragma unroll
      for (int i = 0; i < 8; ++i) {
        int ml = wm * 64 + f * 16 + lhi * 8 + i;
        float w = rw[ml];
        if (w != 0.f) {
          int tok = rt[ml];
          float v = acc[f][g][i] + bias;
          float* p = y + (size_t)tok * D + n;
          *p += w * v;
        }
      }
    }
  }
#undef STAGE_A
#undef WAIT_ASYNC
}

__global__ void zero_kernel(float* __restrict__ y, int n) {
  int i = blockIdx.x * 256 + threadIdx.x;
  if (i < n) y[i] = 0.f;
}

// ---------------- Host launch ----------------
extern "C" void kernel_launch(void* const* d_in, const int* in_sizes, int n_in,
                              void* d_out, int out_size, void* d_ws, size_t ws_size,
                              hipStream_t stream) {
  (void)in_sizes; (void)n_in; (void)out_size; (void)ws_size;
  const float* x      = (const float*)d_in[0];
  const float* gate_w = (const float*)d_in[1];
  const float* w_in   = (const float*)d_in[2];
  const float* b_in   = (const float*)d_in[3];
  const float* w_out  = (const float*)d_in[4];
  const float* b_out  = (const float*)d_in[5];
  const float* sw_in  = (const float*)d_in[6];
  const float* sb_in  = (const float*)d_in[7];
  const float* sw_out = (const float*)d_in[8];
  const float* sb_out = (const float*)d_in[9];
  float* y = (float*)d_out;

  char* ws = (char*)d_ws;
  size_t cur = 0;
  auto alloc = [&](size_t bytes) -> void* {
    void* p = ws + cur;
    cur += (bytes + 255) & ~(size_t)255;
    return p;
  };
  float*          w_tok   = (float*)alloc((size_t)T * E * sizeof(float));
  int*            segoff  = (int*)alloc(16 * sizeof(int));
  int*            tileseg = (int*)alloc(MTILES * sizeof(int));
  int*            rowtok  = (int*)alloc((size_t)RCAP * sizeof(int));
  float*          roww    = (float*)alloc((size_t)RCAP * sizeof(float));
  unsigned short* hbuf    = (unsigned short*)alloc((size_t)RCAP * H * sizeof(unsigned short));

  gate_kernel<<<T / 8, 256, 0, stream>>>(x, gate_w, w_tok);
  build_kernel<<<1, 256, 0, stream>>>(w_tok, segoff, tileseg, rowtok, roww);
  gemm1_kernel<<<dim3(MTILES, H / 128), 256, 0, stream>>>(
      x, w_in, sw_in, b_in, sb_in, rowtok, tileseg, hbuf);
  int ny = T * D;
  zero_kernel<<<(ny + 255) / 256, 256, 0, stream>>>(y, ny);
  for (int s = 0; s < E + 1; ++s) {
    gemm2_kernel<<<dim3(T / 128, D / 128), 256, 0, stream>>>(
        hbuf, w_out, sw_out, b_out, sb_out, segoff, rowtok, roww, y, s);
  }
}
